// DirectForce_14130442404364
// MI455X (gfx1250) — matrix operand order repaired and checked by reading, not compiled, not run on hardware
//
#include <hip/hip_runtime.h>
#include <hip/hip_bf16.h>

typedef __attribute__((ext_vector_type(16))) __bf16 v16bf;
typedef __attribute__((ext_vector_type(8)))  __bf16 v8bf;
typedef __attribute__((ext_vector_type(8)))  float  v8f;
typedef __attribute__((ext_vector_type(4)))  unsigned int u32x4;
typedef __attribute__((ext_vector_type(8)))  int  i32x8;
typedef __attribute__((ext_vector_type(4)))  int  i32x4;

#define E_TOT 262144
#define EH    131072
#define DD    512
#define NATM  4096

__device__ __forceinline__ float ssp_f(float x) {
    // softplus(x) - ln(2), numerically stable
    return fmaxf(x, 0.0f) + log1pf(__expf(-fabsf(x))) - 0.69314718055994531f;
}

// ---- Tensor Data Mover: 2-D tile (bf16) global -> LDS with row padding ----
// D# group0: count=1 | lds_addr | global_addr | type=2
// D# group1: data_size=2B, pad_enable, pad_interval/amount, dims/strides
__device__ __forceinline__ void tdm_load_2d(
    unsigned lds_off, unsigned long long gaddr,
    unsigned tensor_d0, unsigned tensor_d1, unsigned stride0,
    unsigned tile_d0, unsigned tile_d1,
    unsigned pad_interval_code, unsigned pad_amount_code)
{
    u32x4 g0;
    g0.x = 1u;                                            // count=1, user desc
    g0.y = lds_off;                                       // LDS byte address
    g0.z = (unsigned)(gaddr & 0xFFFFFFFFull);             // global_addr[31:0]
    g0.w = (unsigned)((gaddr >> 32) & 0x01FFFFFFull)      // global_addr[56:32]
         | (2u << 30);                                    // type=2 ("image")
    i32x8 g1;
    g1[0] = (int)((1u << 16)                              // data_size = 2 bytes
                | (1u << 20)                              // pad_enable
                | (pad_interval_code << 22)
                | (pad_amount_code << 25));
    g1[1] = (int)((tensor_d0 & 0xFFFFu) << 16);           // tensor_dim0[15:0]
    g1[2] = (int)(((tensor_d0 >> 16) & 0xFFFFu)
                | ((tensor_d1 & 0xFFFFu) << 16));         // dim0 hi | dim1 lo
    g1[3] = (int)(((tensor_d1 >> 16) & 0xFFFFu)
                | ((tile_d0 & 0xFFFFu) << 16));           // dim1 hi | tile0
    g1[4] = (int)(tile_d1 & 0xFFFFu);                     // tile1 | tile2=0
    g1[5] = (int)stride0;                                 // dim0_stride[31:0]
    g1[6] = 0;
    g1[7] = 0;
    i32x4 z4 = {0, 0, 0, 0};
    i32x8 z8 = {0, 0, 0, 0, 0, 0, 0, 0};
    __builtin_amdgcn_tensor_load_to_lds(g0, g1, z4, z4, z8, 0);
}

// ---- prep: W1,W2 -> bf16 transposed [n][k]; zero forces out ----
__global__ __launch_bounds__(256) void k_prep(
    const float* __restrict__ W1, const float* __restrict__ W2,
    __bf16* __restrict__ W1t, __bf16* __restrict__ W2t,
    float* __restrict__ outF)
{
    const int tid = blockIdx.x * 256 + threadIdx.x;   // [0, 512*512)
    const int n = tid >> 9;
    const int k = tid & 511;
    W1t[tid] = (__bf16)W1[k * DD + n];
    W2t[tid] = (__bf16)W2[k * DD + n];
    if (tid < NATM * 3) outF[tid] = 0.0f;
}

// ================= GEMM1: H = ssp(A(f32) @ W1t + b1) =================
// Block owns 128 rows x all 512 cols; A panel staged once in LDS (read A 1x
// from HBM); B tiles (128x32) streamed from L2 by the TDM, double-buffered.
__global__ __launch_bounds__(256) void k_gemm1(
    const float* __restrict__ A, const __bf16* __restrict__ Bt,
    const float* __restrict__ bias, __bf16* __restrict__ Hout)
{
    __shared__ __bf16 As[128][520];      // 512 + 8 pad  (130 KB)
    __shared__ __bf16 Bs[2][128][40];    // 32 + 8 pad, double buffered
    const int tid  = threadIdx.x;
    const int lane = tid & 31;
    const int wave = tid >> 5;
    const int wm   = wave & 3;              // M offset 32*wm
    const int wn   = wave >> 2;             // N offset 64*wn (within n-tile)
    const int m0   = blockIdx.x * 128;
    const int lr   = tid >> 1;
    const int lc   = (tid & 1) << 4;
    const int lrow = lane & 15;
    const int klo  = (lane >> 4) << 3;      // A frag K chunk: 0 or 8
    const int kb   = (lane >> 4) << 4;      // unused; B frags also use klo

    // stage A row-panel fp32 -> bf16 into LDS (once per block)
    for (int kc = 0; kc < DD; kc += 32) {
        const float4* ap = (const float4*)(A + (size_t)(m0 + lr) * DD + (kc + lc));
        float4 a0 = ap[0], a1 = ap[1], a2 = ap[2], a3 = ap[3];
        __bf16* as = &As[lr][kc + lc];
        as[0]=(__bf16)a0.x;  as[1]=(__bf16)a0.y;  as[2]=(__bf16)a0.z;  as[3]=(__bf16)a0.w;
        as[4]=(__bf16)a1.x;  as[5]=(__bf16)a1.y;  as[6]=(__bf16)a1.z;  as[7]=(__bf16)a1.w;
        as[8]=(__bf16)a2.x;  as[9]=(__bf16)a2.y;  as[10]=(__bf16)a2.z; as[11]=(__bf16)a2.w;
        as[12]=(__bf16)a3.x; as[13]=(__bf16)a3.y; as[14]=(__bf16)a3.z; as[15]=(__bf16)a3.w;
    }

    const unsigned bs_off0 = (unsigned)(uintptr_t)&Bs[0][0][0];
    const unsigned bs_off1 = (unsigned)(uintptr_t)&Bs[1][0][0];
    const unsigned long long bbase = (unsigned long long)(uintptr_t)Bt;

    if (wave == 0) {  // tile 0: n-rows [0,128), K [0,32)
        tdm_load_2d(bs_off0, bbase, DD, DD, DD, 32, 128, 3, 3);
        __builtin_amdgcn_s_wait_tensorcnt(0);
    }
    __syncthreads();

    for (int nt = 0; nt < 4; ++nt) {
        v8f acc[2][4] = {};
        for (int kt = 0; kt < 16; ++kt) {
            const int t   = nt * 16 + kt;
            const int buf = t & 1;
            if (wave == 0 && t + 1 < 64) {      // prefetch next B tile via TDM
                const int t1 = t + 1;
                const unsigned long long ga = bbase +
                    (unsigned long long)((unsigned)((t1 >> 4) * 128) * DD
                                         + (unsigned)(t1 & 15) * 32u) * 2ull;
                tdm_load_2d(buf ? bs_off0 : bs_off1, ga, DD, DD, DD, 32, 128, 3, 3);
            }
            const int kk = kt * 32;
            v16bf afr[2], bfr[4];
            #pragma unroll
            for (int mi = 0; mi < 2; ++mi) {
                const v8bf lo = *(const v8bf*)&As[wm*32 + mi*16 + lrow][kk + klo];
                const v8bf hi = *(const v8bf*)&As[wm*32 + mi*16 + lrow][kk + klo + 16];
                afr[mi] = __builtin_shufflevector(lo, hi, 0,1,2,3,4,5,6,7,8,9,10,11,12,13,14,15);
            }
            #pragma unroll
            for (int ni = 0; ni < 4; ++ni) {
                const v8bf lo = *(const v8bf*)&Bs[buf][wn*64 + ni*16 + lrow][klo];
                const v8bf hi = *(const v8bf*)&Bs[buf][wn*64 + ni*16 + lrow][klo + 16];
                bfr[ni] = __builtin_shufflevector(lo, hi, 0,1,2,3,4,5,6,7,8,9,10,11,12,13,14,15);
            }
            #pragma unroll
            for (int mi = 0; mi < 2; ++mi)
                #pragma unroll
                for (int ni = 0; ni < 4; ++ni)
                    acc[mi][ni] = __builtin_amdgcn_wmma_f32_16x16x32_bf16(
                        false, afr[mi], false, bfr[ni], (short)0, acc[mi][ni], false, false);
            if (wave == 0) __builtin_amdgcn_s_wait_tensorcnt(0);
            __syncthreads();
        }
        #pragma unroll
        for (int mi = 0; mi < 2; ++mi) {
            #pragma unroll
            for (int ni = 0; ni < 4; ++ni) {
                const int col = nt*128 + wn*64 + ni*16 + lrow;
                const float bv = bias[col];
                #pragma unroll
                for (int j = 0; j < 8; ++j) {
                    const int row = m0 + wm*32 + mi*16 + ((lane >> 4) << 3) + j;
                    Hout[(size_t)row * DD + col] = (__bf16)ssp_f(acc[mi][ni][j] + bv);
                }
            }
        }
    }
}

// ====== GEMM2 fused with W3 dot: mag[row] = sum_n ssp(H@W2+b2)[row,n]*W3[n] ======
__global__ __launch_bounds__(256) void k_gemm2(
    const __bf16* __restrict__ A, const __bf16* __restrict__ Bt,
    const float* __restrict__ bias, const float* __restrict__ W3,
    float* __restrict__ mag)
{
    __shared__ __bf16 As[128][520];
    __shared__ __bf16 Bs[2][128][40];
    __shared__ float rowsum[128];
    const int tid  = threadIdx.x;
    const int lane = tid & 31;
    const int wave = tid >> 5;
    const int wm   = wave & 3;
    const int wn   = wave >> 2;
    const int m0   = blockIdx.x * 128;
    const int lrow = lane & 15;
    const int klo  = (lane >> 4) << 3;
    const int kb   = (lane >> 4) << 4;

    if (tid < 128) rowsum[tid] = 0.0f;

    const unsigned as_off  = (unsigned)(uintptr_t)&As[0][0];
    const unsigned bs_off0 = (unsigned)(uintptr_t)&Bs[0][0][0];
    const unsigned bs_off1 = (unsigned)(uintptr_t)&Bs[1][0][0];
    const unsigned long long abase = (unsigned long long)(uintptr_t)A;
    const unsigned long long bbase = (unsigned long long)(uintptr_t)Bt;

    if (wave == 0) {
        // A panel: 128 rows x 512 bf16, pad 4 dwords every 256 dwords -> [128][520]
        tdm_load_2d(as_off, abase + (unsigned long long)m0 * DD * 2ull,
                    DD, E_TOT, DD, DD, 128, 7, 3);
        // first B tile
        tdm_load_2d(bs_off0, bbase, DD, DD, DD, 32, 128, 3, 3);
        __builtin_amdgcn_s_wait_tensorcnt(0);
    }
    __syncthreads();

    for (int nt = 0; nt < 4; ++nt) {
        v8f acc[2][4] = {};
        for (int kt = 0; kt < 16; ++kt) {
            const int t   = nt * 16 + kt;
            const int buf = t & 1;
            if (wave == 0 && t + 1 < 64) {
                const int t1 = t + 1;
                const unsigned long long ga = bbase +
                    (unsigned long long)((unsigned)((t1 >> 4) * 128) * DD
                                         + (unsigned)(t1 & 15) * 32u) * 2ull;
                tdm_load_2d(buf ? bs_off0 : bs_off1, ga, DD, DD, DD, 32, 128, 3, 3);
            }
            const int kk = kt * 32;
            v16bf afr[2], bfr[4];
            #pragma unroll
            for (int mi = 0; mi < 2; ++mi) {
                const v8bf lo = *(const v8bf*)&As[wm*32 + mi*16 + lrow][kk + klo];
                const v8bf hi = *(const v8bf*)&As[wm*32 + mi*16 + lrow][kk + klo + 16];
                afr[mi] = __builtin_shufflevector(lo, hi, 0,1,2,3,4,5,6,7,8,9,10,11,12,13,14,15);
            }
            #pragma unroll
            for (int ni = 0; ni < 4; ++ni) {
                const v8bf lo = *(const v8bf*)&Bs[buf][wn*64 + ni*16 + lrow][klo];
                const v8bf hi = *(const v8bf*)&Bs[buf][wn*64 + ni*16 + lrow][klo + 16];
                bfr[ni] = __builtin_shufflevector(lo, hi, 0,1,2,3,4,5,6,7,8,9,10,11,12,13,14,15);
            }
            #pragma unroll
            for (int mi = 0; mi < 2; ++mi)
                #pragma unroll
                for (int ni = 0; ni < 4; ++ni)
                    acc[mi][ni] = __builtin_amdgcn_wmma_f32_16x16x32_bf16(
                        false, afr[mi], false, bfr[ni], (short)0, acc[mi][ni], false, false);
            if (wave == 0) __builtin_amdgcn_s_wait_tensorcnt(0);
            __syncthreads();
        }
        #pragma unroll
        for (int mi = 0; mi < 2; ++mi) {
            #pragma unroll
            for (int ni = 0; ni < 4; ++ni) {
                const int col = nt*128 + wn*64 + ni*16 + lrow;
                const float bv = bias[col];
                const float w3 = W3[col];
                #pragma unroll
                for (int j = 0; j < 8; ++j) {
                    const int rloc = wm*32 + mi*16 + ((lane >> 4) << 3) + j;
                    atomicAdd(&rowsum[rloc], ssp_f(acc[mi][ni][j] + bv) * w3);
                }
            }
        }
    }
    __syncthreads();
    if (tid < 128) mag[m0 + tid] = rowsum[tid];   // block owns full N: no global atomic
}

// ---- force scatter: partner of edge e is e +/- E/2 (structural reverse) ----
__global__ __launch_bounds__(256) void k_force(
    const float* __restrict__ mag, const float* __restrict__ vec,
    const float* __restrict__ len, const int* __restrict__ eidx,
    const float* __restrict__ b3, float* __restrict__ out)
{
    const int e = blockIdx.x * 256 + threadIdx.x;
    if (e >= E_TOT) return;
    const int p = (e < EH) ? (e + EH) : (e - EH);
    const float m = 0.5f * (mag[e] + mag[p]) + b3[0];
    const float s = m / len[e];
    const int c = eidx[e];
    atomicAdd(&out[c * 3 + 0], vec[e * 3 + 0] * s);
    atomicAdd(&out[c * 3 + 1], vec[e * 3 + 1] * s);
    atomicAdd(&out[c * 3 + 2], vec[e * 3 + 2] * s);
}

extern "C" void kernel_launch(void* const* d_in, const int* in_sizes, int n_in,
                              void* d_out, int out_size, void* d_ws, size_t ws_size,
                              hipStream_t stream) {
    const float* edge_emb     = (const float*)d_in[0];
    const float* edge_vectors = (const float*)d_in[1];
    const float* edge_lengths = (const float*)d_in[2];
    const float* W1 = (const float*)d_in[3];
    const float* b1 = (const float*)d_in[4];
    const float* W2 = (const float*)d_in[5];
    const float* b2 = (const float*)d_in[6];
    const float* W3 = (const float*)d_in[7];
    const float* b3 = (const float*)d_in[8];
    const int* edge_index = (const int*)d_in[9];

    char* ws = (char*)d_ws;
    __bf16* W1t = (__bf16*)(ws);
    __bf16* W2t = (__bf16*)(ws + (size_t)DD * DD * 2);
    float*  mag = (float*)(ws + (size_t)DD * DD * 4);
    __bf16* H1  = (__bf16*)(ws + (size_t)DD * DD * 4 + (size_t)E_TOT * 4);

    k_prep<<<E_TOT / 256, 256, 0, stream>>>(W1, W2, W1t, W2t, (float*)d_out);

    k_gemm1<<<E_TOT / 128, 256, 0, stream>>>(edge_emb, W1t, b1, H1);
    k_gemm2<<<E_TOT / 128, 256, 0, stream>>>(H1, W2t, b2, W3, mag);

    k_force<<<E_TOT / 256, 256, 0, stream>>>(mag, edge_vectors, edge_lengths,
                                             edge_index, b3, (float*)d_out);
}
